// TransformerBlock_64699387347185
// MI455X (gfx1250) — matrix operand order repaired and checked by reading, not compiled, not run on hardware
//
#include <hip/hip_runtime.h>
#include <hip/hip_bf16.h>

// ---------------------------------------------------------------------------
// Types for CDNA5 WMMA (wave32): bf16 16x16x32, f32 accumulate.
// ---------------------------------------------------------------------------
typedef __attribute__((ext_vector_type(16))) __bf16 bf16x16;
typedef __attribute__((ext_vector_type(8)))  __bf16 bf16x8;
typedef __attribute__((ext_vector_type(8)))  float  f32x8;

__device__ __forceinline__ __bf16 f2bf(float f) { return (__bf16)f; }
__device__ __forceinline__ float  bf2f(__bf16 b) { return (float)b; }

__device__ __forceinline__ f32x8 zero8() {
  f32x8 z;
#pragma unroll
  for (int i = 0; i < 8; ++i) z[i] = 0.0f;
  return z;
}

// A-fragment (16-bit A, 16x32, §7.12.2): lane = (l16, hf); per lane two
// contiguous 8-elem runs: K = hf*8 + e (e<8) and K = 16 + hf*8 + (e-8).
__device__ __forceinline__ bf16x16 frag_a(const __bf16* row, int k, int hf) {
  bf16x8 lo = *(const bf16x8*)(row + k + hf * 8);
  bf16x8 hi = *(const bf16x8*)(row + k + 16 + hf * 8);
  bf16x16 r;
#pragma unroll
  for (int i = 0; i < 8; ++i) { r[i] = lo[i]; r[i + 8] = hi[i]; }
  return r;
}

// B-fragment (16-bit B, 32x16): lane l16 = N, K = hf*8 + e (e<8) and K = 16 + hf*8 + (e-8).
__device__ __forceinline__ bf16x16 frag_b(const __bf16* row, int k, int hf) {
  bf16x8 lo = *(const bf16x8*)(row + k + hf * 8);
  bf16x8 hi = *(const bf16x8*)(row + k + 16 + hf * 8);
  bf16x16 r;
#pragma unroll
  for (int i = 0; i < 8; ++i) { r[i] = lo[i]; r[i + 8] = hi[i]; }
  return r;
}

__device__ __forceinline__ f32x8 wmma_bf16(bf16x16 a, bf16x16 b, f32x8 c) {
  return __builtin_amdgcn_wmma_f32_16x16x32_bf16(false, a, false, b,
                                                 (short)0, c, false, false);
}

// ---------------------------------------------------------------------------
// Async global->LDS staging (GLOBAL_LOAD_ASYNC_TO_LDS_B128, ASYNCcnt).
// Per-lane: VDST = LDS byte address VGPR, VADDR = 64-bit global address.
// ---------------------------------------------------------------------------
__device__ __forceinline__ void async_b128_to_lds(__bf16* lds_dst, const void* g) {
  const unsigned lds = (unsigned)(size_t)lds_dst;  // addr[31:0] = LDS offset
  asm volatile("global_load_async_to_lds_b128 %0, %1, off"
               :: "v"(lds), "v"(g) : "memory");
}

template <int N>
__device__ __forceinline__ void wait_asynccnt() {
  asm volatile("s_wait_asynccnt %0" :: "i"(N) : "memory");
}

// Raw workgroup split-barrier WITHOUT the counter fence of __syncthreads():
// completion of our async B copies is already guaranteed by s_wait_asynccnt,
// and completion of our ds_loads is guaranteed by WMMA register dataflow.
// This keeps in-flight A-fragment global loads alive across the barrier.
__device__ __forceinline__ void wg_barrier() {
  asm volatile("s_barrier_signal -1\n\ts_barrier_wait -1" ::: "memory");
}

// Stage a 64(N) x 32(K) bf16 tile: 256 threads x one b128 each.
__device__ __forceinline__ void stage_b(__bf16* dst, const __bf16* src, int ldb,
                                        int tid) {
  const int row = tid >> 2;            // 0..63
  const int seg = (tid & 3) * 8;       // 0,8,16,24
  async_b128_to_lds(dst + row * 32 + seg, src + (size_t)row * ldb + seg);
}

// ---------------------------------------------------------------------------
// Cooperative block GEMM core: 8 waves, block tile = 128(M) x 64(N).
// B double-buffered in LDS via async copies (ASYNCcnt); A double-buffered in
// registers so WMMAs never wait on same-iteration global loads.
// ---------------------------------------------------------------------------
template <int KDIM>
__device__ __forceinline__ void gemm_block_16x64(const __bf16* __restrict__ arow,
                                                 const __bf16* __restrict__ Bsrc,
                                                 int ldb, int tid, int l16, int hf,
                                                 f32x8 acc[4]) {
  __shared__ __bf16 Bt[2][64 * 32];
  stage_b(Bt[0], Bsrc, ldb, tid);
  bf16x16 a_cur = frag_a(arow, 0, hf);
  int cur = 0;
  for (int k = 0; k < KDIM; k += 32) {
    const bool more = (k + 32) < KDIM;
    bf16x16 a_nxt;
    if (more) {                              // prefetch next chunk: B -> LDS buf,
      stage_b(Bt[cur ^ 1], Bsrc + k + 32, ldb, tid);
      a_nxt = frag_a(arow, k + 32, hf);      // A -> registers (stays in flight)
      __builtin_prefetch(arow + k + 512, 0, 3);
      wait_asynccnt<1>();                    // current B done, next B in flight
    } else {
      wait_asynccnt<0>();
    }
    wg_barrier();                            // all waves' B copies visible
#pragma unroll
    for (int f = 0; f < 4; ++f) {
      bf16x16 b = frag_b(Bt[cur] + (f * 16 + l16) * 32, 0, hf);
      acc[f] = wmma_bf16(a_cur, b, acc[f]);
    }
    wg_barrier();                            // all reads done before buf reuse
    if (more) a_cur = a_nxt;
    cur ^= 1;
  }
}

// ---------------------------------------------------------------------------
// 0) fp32 -> bf16 weight conversion (w1/w2 become 134MB total: L2-resident)
// ---------------------------------------------------------------------------
__global__ void k_cvt_bf16(const float* __restrict__ in, __bf16* __restrict__ out,
                           long n) {
  long i = (long)blockIdx.x * blockDim.x + threadIdx.x;
  long stride = (long)gridDim.x * blockDim.x;
  for (; i < n; i += stride) out[i] = f2bf(in[i]);
}

// ---------------------------------------------------------------------------
// 1) RMSNorm: one block per token row (D=1024), output bf16
// ---------------------------------------------------------------------------
__global__ void k_rmsnorm(const float* __restrict__ x, const float* __restrict__ w,
                          __bf16* __restrict__ out) {
  const int row = blockIdx.x, tid = threadIdx.x;
  const float4 v = ((const float4*)(x + (size_t)row * 1024))[tid];
  float ss = v.x * v.x + v.y * v.y + v.z * v.z + v.w * v.w;
#pragma unroll
  for (int off = 16; off; off >>= 1) ss += __shfl_xor(ss, off);
  __shared__ float sh[8];
  if ((tid & 31) == 0) sh[tid >> 5] = ss;
  __syncthreads();
  float tot = 0.0f;
#pragma unroll
  for (int i = 0; i < 8; ++i) tot += sh[i];
  const float inv = rsqrtf(tot * (1.0f / 1024.0f) + 1.1920929e-7f);
  const float4 wv = ((const float4*)w)[tid];
  __bf16* o = out + (size_t)row * 1024 + tid * 4;
  o[0] = f2bf(v.x * inv * wv.x);
  o[1] = f2bf(v.y * inv * wv.y);
  o[2] = f2bf(v.z * inv * wv.z);
  o[3] = f2bf(v.w * inv * wv.w);
}

// ---------------------------------------------------------------------------
// 2) QKV projection: [2048,1024]bf16 @ [3072,1024]^T -> f32 [2048,3072]
//    grid (S/128, 3072/64), block 256 (8 waves = 128 M rows)
// ---------------------------------------------------------------------------
__global__ void k_gemm_qkv(const __bf16* __restrict__ A, const __bf16* __restrict__ B,
                           float* __restrict__ C) {
  const int tid = threadIdx.x, lane = tid & 31, wave = tid >> 5;
  const int l16 = lane & 15, hf = lane >> 4;
  const int m0 = blockIdx.x * 128 + wave * 16;
  const int n0 = blockIdx.y * 64;
  const __bf16* arow = A + (size_t)(m0 + l16) * 1024;
  f32x8 acc[4] = {zero8(), zero8(), zero8(), zero8()};
  gemm_block_16x64<1024>(arow, B + (size_t)n0 * 1024, 1024, tid, l16, hf, acc);
#pragma unroll
  for (int f = 0; f < 4; ++f)
#pragma unroll
    for (int r = 0; r < 8; ++r)
      C[(size_t)(m0 + r + 8 * hf) * 3072 + n0 + f * 16 + l16] = acc[f][r];
}

// ---------------------------------------------------------------------------
// 3) RoPE on q/k (f32 math) -> bf16 [H][S][DK]; v -> bf16 transposed [H][DK][S]
// ---------------------------------------------------------------------------
__global__ void k_rope(const float* __restrict__ qkv, __bf16* __restrict__ qr,
                       __bf16* __restrict__ kr, __bf16* __restrict__ vt) {
  const int s = blockIdx.x;
  const float* row = qkv + (size_t)s * 3072;
  for (int p = threadIdx.x; p < 512; p += 256) {
    const int h = p >> 5, i = p & 31;
    const float invf = __powf(10000.0f, -(float)(2 * i) * (1.0f / 64.0f));
    float sn, cs;
    __sincosf((float)s * invf, &sn, &cs);
    const size_t base = ((size_t)h * 2048 + s) * 64;
    const float q1 = row[h * 64 + i], q2 = row[h * 64 + i + 32];
    qr[base + i] = f2bf(q1 * cs - q2 * sn);
    qr[base + i + 32] = f2bf(q2 * cs + q1 * sn);
    const float k1 = row[1024 + h * 64 + i], k2 = row[1024 + h * 64 + i + 32];
    kr[base + i] = f2bf(k1 * cs - k2 * sn);
    kr[base + i + 32] = f2bf(k2 * cs + k1 * sn);
  }
  for (int p = threadIdx.x; p < 1024; p += 256) {
    const int h = p >> 6, d = p & 63;
    vt[((size_t)h * 64 + d) * 2048 + s] = f2bf(row[2048 + h * 64 + d]);
  }
}

// ---------------------------------------------------------------------------
// 4) Flash attention: one wave per (head, 16-query tile). Causal, online
//    softmax; P staged through LDS row-major so it reloads as an A-fragment.
// ---------------------------------------------------------------------------
__global__ void __launch_bounds__(32) k_flash(const __bf16* __restrict__ qr,
                                              const __bf16* __restrict__ kr,
                                              const __bf16* __restrict__ vt,
                                              __bf16* __restrict__ attn) {
  const int qt = blockIdx.x, h = blockIdx.y;
  const int lane = threadIdx.x & 31, l16 = lane & 15, hf = lane >> 4;
  __shared__ __bf16 P[16][32];

  const __bf16* qrow = qr + ((size_t)h * 2048 + qt * 16 + l16) * 64;
  const bf16x16 qa0 = frag_a(qrow, 0, hf);
  const bf16x16 qa1 = frag_a(qrow, 32, hf);

  f32x8 acc[4] = {zero8(), zero8(), zero8(), zero8()};
  float m[8], l[8];
#pragma unroll
  for (int r = 0; r < 8; ++r) { m[r] = -1e30f; l[r] = 0.0f; }

  const int jpEnd = (qt * 16 + 15) / 32;  // inclusive; 32 keys per step
  for (int jp = 0; jp <= jpEnd; ++jp) {
    f32x8 s0 = zero8(), s1 = zero8();
    const __bf16* krow0 = kr + ((size_t)h * 2048 + jp * 32 + l16) * 64;
    const __bf16* krow1 = krow0 + 16 * 64;
    s0 = wmma_bf16(qa0, frag_b(krow0, 0, hf), s0);
    s0 = wmma_bf16(qa1, frag_b(krow0, 32, hf), s0);
    s1 = wmma_bf16(qa0, frag_b(krow1, 0, hf), s1);
    s1 = wmma_bf16(qa1, frag_b(krow1, 32, hf), s1);

#pragma unroll
    for (int r = 0; r < 8; ++r) {
      const int row = qt * 16 + r + 8 * hf;
      const int c0 = jp * 32 + l16, c1 = c0 + 16;
      float v0 = (c0 <= row) ? s0[r] * 0.125f : -1e30f;   // 1/sqrt(64)
      float v1 = (c1 <= row) ? s1[r] * 0.125f : -1e30f;
      float mx = fmaxf(v0, v1);
#pragma unroll
      for (int off = 8; off; off >>= 1) mx = fmaxf(mx, __shfl_xor(mx, off));
      const float nm = fmaxf(m[r], mx);
      const float corr = __expf(m[r] - nm);
      const float p0 = (v0 > -1e29f) ? __expf(v0 - nm) : 0.0f;
      const float p1 = (v1 > -1e29f) ? __expf(v1 - nm) : 0.0f;
      float rs = p0 + p1;
#pragma unroll
      for (int off = 8; off; off >>= 1) rs += __shfl_xor(rs, off);
      l[r] = l[r] * corr + rs;
      m[r] = nm;
#pragma unroll
      for (int f = 0; f < 4; ++f) acc[f][r] *= corr;
      P[r + 8 * hf][l16] = f2bf(p0);
      P[r + 8 * hf][16 + l16] = f2bf(p1);
    }
    __syncthreads();
    const bf16x16 pa = frag_a(&P[l16][0], 0, hf);  // ds_load_b128 x2
#pragma unroll
    for (int f = 0; f < 4; ++f) {
      const __bf16* vrow = vt + ((size_t)h * 64 + f * 16 + l16) * 2048 + jp * 32;
      acc[f] = wmma_bf16(pa, frag_b(vrow, 0, hf), acc[f]);
    }
    __syncthreads();
  }
#pragma unroll
  for (int r = 0; r < 8; ++r) {
    const float inv = 1.0f / l[r];
#pragma unroll
    for (int f = 0; f < 4; ++f)
      attn[(size_t)(qt * 16 + r + 8 * hf) * 1024 + h * 64 + f * 16 + l16] =
          f2bf(acc[f][r] * inv);
  }
}

// ---------------------------------------------------------------------------
// 5) Out projection + residual: xmid = x + attn @ out_w^T   grid (16, 16)
// ---------------------------------------------------------------------------
__global__ void k_outproj(const __bf16* __restrict__ A, const __bf16* __restrict__ B,
                          const float* __restrict__ x, float* __restrict__ xmid) {
  const int tid = threadIdx.x, lane = tid & 31, wave = tid >> 5;
  const int l16 = lane & 15, hf = lane >> 4;
  const int m0 = blockIdx.x * 128 + wave * 16;
  const int n0 = blockIdx.y * 64;
  const __bf16* arow = A + (size_t)(m0 + l16) * 1024;
  f32x8 acc[4] = {zero8(), zero8(), zero8(), zero8()};
  gemm_block_16x64<1024>(arow, B + (size_t)n0 * 1024, 1024, tid, l16, hf, acc);
#pragma unroll
  for (int f = 0; f < 4; ++f)
#pragma unroll
    for (int r = 0; r < 8; ++r) {
      const size_t idx = (size_t)(m0 + r + 8 * hf) * 1024 + n0 + f * 16 + l16;
      xmid[idx] = x[idx] + acc[f][r];
    }
}

// ---------------------------------------------------------------------------
// 6) Router: one wave per token; softmax over 8 experts, top-2, renorm,
//    build per-expert token lists (int atomics only -> deterministic output).
// ---------------------------------------------------------------------------
__global__ void k_init_counts(int* counts) {
  if (threadIdx.x < 8) counts[threadIdx.x] = 0;
}

__global__ void k_router(const __bf16* __restrict__ h2, const float* __restrict__ rw,
                         float* __restrict__ probs, float* __restrict__ top_p,
                         int* __restrict__ counts, int* __restrict__ lists) {
  const int wave = threadIdx.x >> 5, lane = threadIdx.x & 31;
  const int t = blockIdx.x * 8 + wave;
  float acc[8];
#pragma unroll
  for (int e = 0; e < 8; ++e) acc[e] = 0.0f;
  const __bf16* hrow = h2 + (size_t)t * 1024;
  for (int d = lane; d < 1024; d += 32) {
    const float hv = bf2f(hrow[d]);
#pragma unroll
    for (int e = 0; e < 8; ++e) acc[e] += hv * rw[e * 1024 + d];
  }
#pragma unroll
  for (int e = 0; e < 8; ++e)
#pragma unroll
    for (int off = 16; off; off >>= 1) acc[e] += __shfl_xor(acc[e], off);
  if (lane == 0) {
    float mx = acc[0];
#pragma unroll
    for (int e = 1; e < 8; ++e) mx = fmaxf(mx, acc[e]);
    float p[8], sum = 0.0f;
#pragma unroll
    for (int e = 0; e < 8; ++e) { p[e] = __expf(acc[e] - mx); sum += p[e]; }
    const float inv = 1.0f / sum;
#pragma unroll
    for (int e = 0; e < 8; ++e) { p[e] *= inv; probs[t * 8 + e] = p[e]; }
    int i0 = 0;
#pragma unroll
    for (int e = 1; e < 8; ++e) if (p[e] > p[i0]) i0 = e;
    int i1 = (i0 == 0) ? 1 : 0;
#pragma unroll
    for (int e = 0; e < 8; ++e) if (e != i0 && p[e] > p[i1]) i1 = e;
    const float s2 = 1.0f / (p[i0] + p[i1]);
    top_p[t * 2] = p[i0] * s2;
    top_p[t * 2 + 1] = p[i1] * s2;
    const int pos0 = atomicAdd(&counts[i0], 1);
    lists[i0 * 2048 + pos0] = t * 2;        // code = token*2 + slot
    const int pos1 = atomicAdd(&counts[i1], 1);
    lists[i1 * 2048 + pos1] = t * 2 + 1;
  }
}

// ---------------------------------------------------------------------------
// 7) MoE expert GEMM1: hid[code] = gelu(h2[token] @ w1[e]^T), gathered rows
//    grid (16, E, 64). Only block-level early-exit (barriers need all waves).
// ---------------------------------------------------------------------------
__global__ void k_moe1(const __bf16* __restrict__ h2, const __bf16* __restrict__ w1bf,
                       const int* __restrict__ counts, const int* __restrict__ lists,
                       __bf16* __restrict__ hid) {
  const int e = blockIdx.y;
  const int cnt = counts[e];
  if ((int)blockIdx.x * 128 >= cnt) return;         // uniform across block
  const int tid = threadIdx.x, lane = tid & 31, wave = tid >> 5;
  const int l16 = lane & 15, hf = lane >> 4;
  const int m0 = blockIdx.x * 128 + wave * 16;
  const int n0 = blockIdx.z * 64;
  const int idx = m0 + l16;
  const int code = (idx < cnt) ? lists[e * 2048 + idx] : -1;
  const __bf16* arow = h2 + (size_t)(code < 0 ? 0 : (code >> 1)) * 1024;
  f32x8 acc[4] = {zero8(), zero8(), zero8(), zero8()};
  gemm_block_16x64<1024>(arow, w1bf + ((size_t)e * 4096 + n0) * 1024, 1024,
                         tid, l16, hf, acc);
#pragma unroll
  for (int r = 0; r < 8; ++r) {
    const int cm = __shfl(code, r + 8 * hf);  // code of output row M
    if (cm < 0) continue;
#pragma unroll
    for (int f = 0; f < 4; ++f) {
      const float v = acc[f][r];
      const float g = 0.5f * v * (1.0f + erff(v * 0.70710678f));  // exact gelu
      hid[(size_t)cm * 4096 + n0 + f * 16 + l16] = f2bf(g);
    }
  }
}

// ---------------------------------------------------------------------------
// 8) MoE expert GEMM2: eo[code] = hid[code] @ w2[e]^T  (f32), grid (16, E, 16)
// ---------------------------------------------------------------------------
__global__ void k_moe2(const __bf16* __restrict__ hid, const __bf16* __restrict__ w2bf,
                       const int* __restrict__ counts, const int* __restrict__ lists,
                       float* __restrict__ eo) {
  const int e = blockIdx.y;
  const int cnt = counts[e];
  if ((int)blockIdx.x * 128 >= cnt) return;         // uniform across block
  const int tid = threadIdx.x, lane = tid & 31, wave = tid >> 5;
  const int l16 = lane & 15, hf = lane >> 4;
  const int m0 = blockIdx.x * 128 + wave * 16;
  const int n0 = blockIdx.z * 64;
  const int idx = m0 + l16;
  const int code = (idx < cnt) ? lists[e * 2048 + idx] : -1;
  const __bf16* arow = hid + (size_t)(code < 0 ? 0 : code) * 4096;
  f32x8 acc[4] = {zero8(), zero8(), zero8(), zero8()};
  gemm_block_16x64<4096>(arow, w2bf + ((size_t)e * 1024 + n0) * 4096, 4096,
                         tid, l16, hf, acc);
#pragma unroll
  for (int r = 0; r < 8; ++r) {
    const int cm = __shfl(code, r + 8 * hf);
    if (cm < 0) continue;
#pragma unroll
    for (int f = 0; f < 4; ++f)
      eo[(size_t)cm * 1024 + n0 + f * 16 + l16] = acc[f][r];
  }
}

// ---------------------------------------------------------------------------
// 9) Combine: out = xmid + p0*eo[slot0] + p1*eo[slot1]
// ---------------------------------------------------------------------------
__global__ void k_combine(const float* __restrict__ xmid, const float* __restrict__ eo,
                          const float* __restrict__ top_p, float* __restrict__ out) {
  const int t = blockIdx.x, d = threadIdx.x * 4;
  const float4 xv = *(const float4*)(xmid + (size_t)t * 1024 + d);
  const float p0 = top_p[t * 2], p1 = top_p[t * 2 + 1];
  const float4 a = *(const float4*)(eo + (size_t)(t * 2) * 1024 + d);
  const float4 b = *(const float4*)(eo + (size_t)(t * 2 + 1) * 1024 + d);
  float4 r;
  r.x = xv.x + p0 * a.x + p1 * b.x;
  r.y = xv.y + p0 * a.y + p1 * b.y;
  r.z = xv.z + p0 * a.z + p1 * b.z;
  r.w = xv.w + p0 * a.w + p1 * b.w;
  *(float4*)(out + (size_t)t * 1024 + d) = r;
}

// ---------------------------------------------------------------------------
// 10) Aux loss: E * sum_e(mean_t probs)^2, fixed-order (deterministic) tree
// ---------------------------------------------------------------------------
__global__ void k_aux(const float* __restrict__ probs, float* __restrict__ auxOut) {
  const int lane = threadIdx.x & 31, e = threadIdx.x >> 5;  // 8 waves, 8 experts
  float s = 0.0f;
  for (int t = lane; t < 2048; t += 32) s += probs[t * 8 + e];
#pragma unroll
  for (int off = 16; off; off >>= 1) s += __shfl_xor(s, off);
  __shared__ float sh[8];
  if (lane == 0) sh[e] = s * (1.0f / 2048.0f);
  __syncthreads();
  if (threadIdx.x == 0) {
    float a = 0.0f;
#pragma unroll
    for (int i = 0; i < 8; ++i) a += sh[i] * sh[i];
    auxOut[0] = 8.0f * a;
  }
}

// ---------------------------------------------------------------------------
// Host launcher
// ---------------------------------------------------------------------------
extern "C" void kernel_launch(void* const* d_in, const int* in_sizes, int n_in,
                              void* d_out, int out_size, void* d_ws, size_t ws_size,
                              hipStream_t stream) {
  const float* x        = (const float*)d_in[0];
  const float* qkv_w    = (const float*)d_in[1];
  const float* out_w    = (const float*)d_in[2];
  const float* router_w = (const float*)d_in[3];
  const float* w1       = (const float*)d_in[4];
  const float* w2       = (const float*)d_in[5];
  const float* n1w      = (const float*)d_in[6];
  const float* n2w      = (const float*)d_in[7];
  float* out = (float*)d_out;

  char* p = (char*)d_ws;
  auto take = [&](size_t bytes) {
    char* r = p;
    p += (bytes + 255) & ~(size_t)255;
    return r;
  };
  __bf16* h1     = (__bf16*)take((size_t)2048 * 1024 * 2);
  __bf16* qkvwbf = (__bf16*)take((size_t)3072 * 1024 * 2);
  __bf16* outwbf = (__bf16*)take((size_t)1024 * 1024 * 2);
  __bf16* w1bf   = (__bf16*)take((size_t)8 * 4096 * 1024 * 2);
  __bf16* w2bf   = (__bf16*)take((size_t)8 * 1024 * 4096 * 2);
  float*  qkvf   = (float*)take((size_t)2048 * 3072 * 4);
  __bf16* qr     = (__bf16*)take((size_t)16 * 2048 * 64 * 2);
  __bf16* kr     = (__bf16*)take((size_t)16 * 2048 * 64 * 2);
  __bf16* vt     = (__bf16*)take((size_t)16 * 64 * 2048 * 2);
  __bf16* attn   = (__bf16*)take((size_t)2048 * 1024 * 2);
  float*  xmid   = (float*)take((size_t)2048 * 1024 * 4);
  __bf16* h2     = (__bf16*)take((size_t)2048 * 1024 * 2);
  float*  probs  = (float*)take((size_t)2048 * 8 * 4);
  float*  topp   = (float*)take((size_t)2048 * 2 * 4);
  int*    counts = (int*)take(8 * 4);
  int*    lists  = (int*)take((size_t)8 * 2048 * 4);
  __bf16* hid    = (__bf16*)take((size_t)4096 * 4096 * 2);
  float*  eo     = (float*)take((size_t)4096 * 1024 * 4);

  // Weights -> bf16 (w1/w2 shrink to 134MB total: resident in 192MB L2)
  k_cvt_bf16<<<2048, 256, 0, stream>>>(qkv_w, qkvwbf, (long)3072 * 1024);
  k_cvt_bf16<<<2048, 256, 0, stream>>>(out_w, outwbf, (long)1024 * 1024);
  k_cvt_bf16<<<8192, 256, 0, stream>>>(w1, w1bf, (long)8 * 4096 * 1024);
  k_cvt_bf16<<<8192, 256, 0, stream>>>(w2, w2bf, (long)8 * 1024 * 4096);

  k_rmsnorm<<<2048, 256, 0, stream>>>(x, n1w, h1);
  k_gemm_qkv<<<dim3(16, 48), 256, 0, stream>>>(h1, qkvwbf, qkvf);
  k_rope<<<2048, 256, 0, stream>>>(qkvf, qr, kr, vt);
  k_flash<<<dim3(128, 16), 32, 0, stream>>>(qr, kr, vt, attn);
  k_outproj<<<dim3(16, 16), 256, 0, stream>>>(attn, outwbf, x, xmid);
  k_rmsnorm<<<2048, 256, 0, stream>>>(xmid, n2w, h2);
  k_init_counts<<<1, 32, 0, stream>>>(counts);
  k_router<<<256, 256, 0, stream>>>(h2, router_w, probs, topp, counts, lists);
  k_moe1<<<dim3(16, 8, 64), 256, 0, stream>>>(h2, w1bf, counts, lists, hid);
  k_moe2<<<dim3(16, 8, 16), 256, 0, stream>>>(hid, w2bf, counts, lists, eo);
  k_combine<<<2048, 256, 0, stream>>>(xmid, eo, topp, out);
  k_aux<<<1, 256, 0, stream>>>(probs, out + (size_t)2048 * 1024);
}
